// RNN_model_46316927320423
// MI455X (gfx1250) — compile-verified, matrix operand order checked by reading, not run on hardware
//
#include <hip/hip_runtime.h>

typedef __attribute__((ext_vector_type(16))) __bf16 v16bf;
typedef __attribute__((ext_vector_type(8)))  float  v8f;

// 16-byte int vector type matching the async-LDS builtin's parameter type
typedef int vi4 __attribute__((vector_size(16)));
typedef __attribute__((address_space(1))) vi4* as1_vi4p;
typedef __attribute__((address_space(3))) vi4* as3_vi4p;

#if defined(__has_builtin)
# if __has_builtin(__builtin_amdgcn_global_load_async_to_lds_b128)
#  define USE_ASYNC_BUILTIN 1
# endif
# if __has_builtin(__builtin_amdgcn_s_wait_asynccnt)
#  define USE_WAITASYNC_BUILTIN 1
# endif
#endif

namespace {

constexpr int kB  = 64;     // batch
constexpr int kT  = 512;    // timesteps
constexpr int kH  = 512;    // hidden
constexpr int kG  = 2048;   // 4*hidden (gates)
constexpr int kE  = 100;    // decoder width
constexpr int kKS = 16;     // K steps of 32 over kH
constexpr int kNBLK = 16;   // persistent recurrent workgroups
constexpr float kKeepInv = 1.0f / 0.7f;
constexpr float kBnEps   = 1e-5f;

// ---- fragment layout helpers (wave32 WMMA 16x16x32 bf16) -------------------
// A-frag (16x32, M x K):  lane = hi*16 + m (m = M row 0..15, hi = lane>>4)
//   element e (0..15):    k = kbase + ((e>>3)<<4) + hi*8 + (e&7)
// B-frag (32x16, K x N):  mirrored with N in place of M.
// C/D (v8f):              element r: M = r + hi*8, N = lane&15.
// Frags stored pre-swizzled so each lane reads ONE contiguous 32B chunk.
__device__ __forceinline__ long fidx(int nt, int ks, int lane) {
  return (((long)(nt * kKS + ks) * 32) + lane) * 16;
}
__device__ __forceinline__ int koff(int e, int hi) {
  return ((e >> 3) << 4) + hi * 8 + (e & 7);
}
__device__ __forceinline__ v8f wmma_bf16(v16bf a, v16bf b, v8f c) {
  return __builtin_amdgcn_wmma_f32_16x16x32_bf16(false, a, false, b,
                                                 (short)0, c, false, false);
}
__device__ __forceinline__ float sigf(float x) {
  return 1.0f / (1.0f + __expf(-x));
}

// ---- CDNA5 async global->LDS copy (16 bytes per lane), ASYNCcnt tracked ----
__device__ __forceinline__ void async_copy_b128(const void* g, void* l) {
#ifdef USE_ASYNC_BUILTIN
  __builtin_amdgcn_global_load_async_to_lds_b128(
      (as1_vi4p)(uintptr_t)g, (as3_vi4p)(uintptr_t)l, 0, 0);
#else
  unsigned loff = (unsigned)(uintptr_t)l;   // low 32 bits = LDS byte offset
  asm volatile("global_load_async_to_lds_b128 %0, %1, off"
               :: "v"(loff), "v"(g) : "memory");
#endif
}
__device__ __forceinline__ void wait_async0() {
#ifdef USE_WAITASYNC_BUILTIN
  __builtin_amdgcn_s_wait_asynccnt(0);
#else
  asm volatile("s_wait_asynccnt 0x0" ::: "memory");
#endif
}

// ---- device-wide sense-reversing barrier for the persistent kernel --------
__device__ __forceinline__ void grid_barrier(unsigned* bar) {
  __syncthreads();
  if (threadIdx.x == 0) {
    unsigned* cnt = bar;
    unsigned* gen = bar + 1;
    __threadfence();
    unsigned g = __hip_atomic_load(gen, __ATOMIC_RELAXED, __HIP_MEMORY_SCOPE_AGENT);
    unsigned a = __hip_atomic_fetch_add(cnt, 1u, __ATOMIC_ACQ_REL, __HIP_MEMORY_SCOPE_AGENT);
    if (a == (unsigned)(kNBLK - 1)) {
      __hip_atomic_store(cnt, 0u, __ATOMIC_RELAXED, __HIP_MEMORY_SCOPE_AGENT);
      __hip_atomic_fetch_add(gen, 1u, __ATOMIC_RELEASE, __HIP_MEMORY_SCOPE_AGENT);
    } else {
      while (__hip_atomic_load(gen, __ATOMIC_ACQUIRE, __HIP_MEMORY_SCOPE_AGENT) == g) {
        __builtin_amdgcn_s_sleep(8);
      }
    }
    __threadfence();
  }
  __syncthreads();
}

} // namespace

// ============================================================================
// Kernel 1: convert weights to bf16 in WMMA B-fragment layout; init barrier.
// ============================================================================
__global__ void prep_kernel(const float* __restrict__ w_hh,
                            const float* __restrict__ w_ih,
                            const float* __restrict__ dec_w,
                            __bf16* __restrict__ whhf,
                            __bf16* __restrict__ wihf,
                            __bf16* __restrict__ decwf,
                            unsigned* __restrict__ bar) {
  long gid = (long)blockIdx.x * blockDim.x + threadIdx.x;
  if (gid == 0) { bar[0] = 0u; bar[1] = 0u; }
  long stride = (long)gridDim.x * blockDim.x;

  const long NW = (long)(kG / 16) * kKS * 32 * 16;   // 1,048,576 halfs
  for (long i = gid; i < NW; i += stride) {
    int e = i & 15, lane = (i >> 4) & 31, ks = (i >> 9) & 15, nt = (int)(i >> 13);
    int hl = lane & 15, hi = lane >> 4;
    int n = nt * 16 + hl;
    int k = ks * 32 + koff(e, hi);
    // B-matrix element B[k][n] = W[n][k]  (gates = X @ W^T)
    whhf[i] = (__bf16)w_hh[(long)n * kH + k];
    wihf[i] = (__bf16)w_ih[(long)n * kH + k];
  }
  const long ND = (long)7 * kKS * 32 * 16;           // 57,344 halfs
  for (long i = gid; i < ND; i += stride) {
    int e = i & 15, lane = (i >> 4) & 31, ks = (i >> 9) & 15, nt = (int)(i >> 13);
    int hl = lane & 15, hi = lane >> 4;
    int n = nt * 16 + hl;
    int k = ks * 32 + koff(e, hi);
    decwf[i] = (n < kE) ? (__bf16)dec_w[(long)n * kH + k] : (__bf16)0.0f;
  }
}

// ============================================================================
// Kernel 2: fused embedding gather + input-gate GEMM (one big WMMA GEMM):
//   Gin[t*64+b][n] = emb[x[b][t]] . w_ih[n] + b_ih[n] + b_hh[n]
// ============================================================================
__global__ void __launch_bounds__(256)
gin_kernel(const int* __restrict__ x, const float* __restrict__ emb,
           const __bf16* __restrict__ wihf,
           const float* __restrict__ b_ih, const float* __restrict__ b_hh,
           float* __restrict__ gin) {
  const int wave = threadIdx.x >> 5;
  const int lane = threadIdx.x & 31;
  const int hl = lane & 15, hi = lane >> 4;

  const int Rbase = blockIdx.x * 128 + wave * 16;    // A-tile row base
  const int R = Rbase + hl;                          // this lane's A row
  const int t = R >> 6, b = R & 63;                  // R = t*64 + b
  const int token = x[b * kT + t];
  const float* erow = emb + (long)token * kH;
  const int nbase = blockIdx.y * 64;

  v8f acc[4];
#pragma unroll
  for (int j = 0; j < 4; ++j)
    acc[j] = (v8f){0.f, 0.f, 0.f, 0.f, 0.f, 0.f, 0.f, 0.f};

  for (int ks = 0; ks < kKS; ++ks) {
    __builtin_prefetch((const void*)(erow + ks * 32 + 64), 0, 1); // global_prefetch
    v16bf a;
#pragma unroll
    for (int e = 0; e < 16; ++e)
      a[e] = (__bf16)erow[ks * 32 + koff(e, hi)];
#pragma unroll
    for (int j = 0; j < 4; ++j) {
      int nt = (nbase >> 4) + j;
      v16bf bb = *(const v16bf*)(wihf + fidx(nt, ks, lane));
      acc[j] = wmma_bf16(a, bb, acc[j]);
    }
  }
#pragma unroll
  for (int j = 0; j < 4; ++j) {
    int n = nbase + j * 16 + hl;
    float bias = b_ih[n] + b_hh[n];
#pragma unroll
    for (int r = 0; r < 8; ++r) {
      int Row = Rbase + r + hi * 8;
      gin[(long)Row * kG + n] = acc[j][r] + bias;
    }
  }
}

// ============================================================================
// Kernel 3: persistent recurrent scan. 16 blocks x 256 threads (8 waves).
// Block bid owns hidden slice [bid*32, bid*32+32) -> 128 gate columns.
// Per step: async-stage h(t-1) frags to LDS (ASYNCcnt) overlapped with Gin
// C-init loads -> WMMA gates GEMM (A from LDS, B whhf from L2) -> LSTM
// pointwise (c in registers) + per-slice BatchNorm + locked dropout ->
// write h/hd bf16 frags -> ONE grid barrier -> WMMA decoder + running max.
// ============================================================================
__global__ void __launch_bounds__(256, 1)
rnn_kernel(const float* __restrict__ gin,
           const __bf16* __restrict__ whhf,
           const __bf16* __restrict__ decwf,
           const float* __restrict__ gamma, const float* __restrict__ beta,
           const float* __restrict__ dec_b, const float* __restrict__ mask,
           __bf16* __restrict__ hfrag, __bf16* __restrict__ hdfrag,
           float* __restrict__ out, unsigned* __restrict__ bar) {
  __shared__ __bf16 hstage[4 * kKS * 32 * 16];   // 64 KB: h(t-1) frag stage
  __shared__ float gates[kB][128];               // 32 KB (h written back into i-slot)
  __shared__ float mu_s[32], rs_s[32];

  const int tid = threadIdx.x;
  const int bid = blockIdx.x;        // 0..15, owns k-slab ks == bid
  const int wave = tid >> 5;
  const int lane = tid & 31;
  const int hl = lane & 15, hi = lane >> 4;
  const int mt_w = wave & 3;         // wave's M tile (batch rows)
  const int jgrp = (wave >> 2) * 4;  // wave's 4 local N tiles

  // Step-invariant per-thread state: this thread owns (hu = tid&31, b = e*8+wave).
  const int hu_t = tid & 31;
  const float ga = gamma[bid * 32 + hu_t];
  const float be = beta[bid * 32 + hu_t];
  float mreg[8], c_reg[8];
#pragma unroll
  for (int e = 0; e < 8; ++e) {
    int b = e * 8 + wave;
    mreg[e] = mask[b * kH + bid * 32 + hu_t] * kKeepInv;
    c_reg[e] = 0.f;
  }
  // zero h frag buffer 1 (read at t=0); block owns k-slab ks == bid
  for (int i = tid; i < 2048; i += 256) {
    int e = i & 15, ln = (i >> 4) & 31, mt = i >> 9;
    hfrag[fidx(1 * 4 + mt, bid, ln) + e] = (__bf16)0.f;
  }
  grid_barrier(bar);

  for (int t = 0; t < kT; ++t) {
    const int buf = t & 1;
    // ---- async-stage full h(t-1) frag buffer (64 KB) into LDS ----
    {
      const __bf16* gsrc = hfrag + (long)(1 - buf) * (4 * kKS * 32 * 16);
#pragma unroll
      for (int j = 0; j < 16; ++j) {
        int chunk = j * 256 + tid;               // 16B chunks, coalesced
        async_copy_b128(gsrc + chunk * 8, hstage + chunk * 8);
      }
    }
    // ---- C init from Gin (overlaps the async copies) ----
    v8f acc[4];
#pragma unroll
    for (int j = 0; j < 4; ++j) {
      int nl = (jgrp + j) * 16 + hl;                  // local col 0..127
      int gate = nl >> 5;
      int n = gate * 512 + bid * 32 + (nl & 31);      // global gate column
      v8f a;
#pragma unroll
      for (int r = 0; r < 8; ++r) {
        int bb = mt_w * 16 + r + hi * 8;
        a[r] = gin[((long)t * kB + bb) * kG + n];
      }
      acc[j] = a;
    }
    wait_async0();
    __syncthreads();   // hstage ready for all waves; gates[] free for reuse
    // ---- gates = h_prev @ whh^T + Gin ----
    for (int ks = 0; ks < kKS; ++ks) {
      v16bf af = *(const v16bf*)(hstage + fidx(mt_w, ks, lane));  // ds_load
#pragma unroll
      for (int j = 0; j < 4; ++j) {
        int nlb = (jgrp + j) * 16;
        int gate = nlb >> 5;
        int nt_nat = (gate * 512 + bid * 32 + (nlb & 31)) >> 4;
        v16bf bf = *(const v16bf*)(whhf + fidx(nt_nat, ks, lane));
        acc[j] = wmma_bf16(af, bf, acc[j]);
      }
    }
#pragma unroll
    for (int j = 0; j < 4; ++j) {
      int nl = (jgrp + j) * 16 + hl;
#pragma unroll
      for (int r = 0; r < 8; ++r)
        gates[mt_w * 16 + r + hi * 8][nl] = acc[j][r];
    }
    __syncthreads();
    // ---- LSTM pointwise; h written back into the i-gate slot (exclusive) ----
#pragma unroll
    for (int e = 0; e < 8; ++e) {
      int b = e * 8 + wave;
      float iv = gates[b][hu_t],      fv = gates[b][32 + hu_t];
      float gv = gates[b][64 + hu_t], ov = gates[b][96 + hu_t];
      float cn = sigf(fv) * c_reg[e] + sigf(iv) * tanhf(gv);
      c_reg[e] = cn;
      gates[b][hu_t] = sigf(ov) * tanhf(cn);   // h
    }
    __syncthreads();
    // ---- BatchNorm stats over batch (local to slice) ----
    if (tid < 32) {
      float s = 0.f, s2 = 0.f;
      for (int b = 0; b < kB; ++b) { float v = gates[b][tid]; s += v; s2 += v * v; }
      float mu = s * (1.f / kB);
      float var = s2 * (1.f / kB) - mu * mu;
      mu_s[tid] = mu;
      rs_s[tid] = rsqrtf(var + kBnEps);
    }
    __syncthreads();
    // ---- normalize + locked dropout; emit bf16 h/hd in frag layout ----
#pragma unroll
    for (int e = 0; e < 8; ++e) {
      int b = e * 8 + wave;
      float h = gates[b][hu_t];
      float hn = (h - mu_s[hu_t]) * rs_s[hu_t] * ga + be;
      float hd = hn * mreg[e];
      int mt = b >> 4, ml = b & 15;
      int hi2 = (hu_t >> 3) & 1;
      int ee = ((hu_t >> 4) << 3) + (hu_t & 7);
      int ln = hi2 * 16 + ml;
      long p = fidx(buf * 4 + mt, bid, ln) + ee;   // block owns k-slab == bid
      hfrag[p]  = (__bf16)h;
      hdfrag[p] = (__bf16)hd;
    }
    grid_barrier(bar);   // single device-wide sync per step
    // ---- decoder tile + running max ----
    int gw = bid * 8 + wave;                 // 0..127 waves; 28 tile jobs
    if (gw < 28) {
      int mt = gw / 7, nt = gw % 7;
      v8f dacc = (v8f){0.f, 0.f, 0.f, 0.f, 0.f, 0.f, 0.f, 0.f};
      for (int ks = 0; ks < kKS; ++ks) {
        v16bf af = *(const v16bf*)(hdfrag + fidx(buf * 4 + mt, ks, lane));
        v16bf bf = *(const v16bf*)(decwf + fidx(nt, ks, lane));
        dacc = wmma_bf16(af, bf, dacc);
      }
      int n = nt * 16 + hl;
      if (n < kE) {
        float db = dec_b[n];
#pragma unroll
        for (int r = 0; r < 8; ++r) {
          int b = mt * 16 + r + hi * 8;
          float v = dacc[r] + db;
          float* op = out + (long)b * kE + n;
          if (t == 0) *op = v;
          else { float o = *op; *op = (v > o) ? v : o; }
        }
      }
    }
  }
}

// ============================================================================
extern "C" void kernel_launch(void* const* d_in, const int* in_sizes, int n_in,
                              void* d_out, int out_size, void* d_ws, size_t ws_size,
                              hipStream_t stream) {
  const int*   x     = (const int*)d_in[0];
  const float* emb   = (const float*)d_in[1];
  const float* w_ih  = (const float*)d_in[2];
  const float* w_hh  = (const float*)d_in[3];
  const float* b_ih  = (const float*)d_in[4];
  const float* b_hh  = (const float*)d_in[5];
  const float* gamma = (const float*)d_in[6];
  const float* beta  = (const float*)d_in[7];
  const float* dec_w = (const float*)d_in[8];
  const float* dec_b = (const float*)d_in[9];
  const float* mask  = (const float*)d_in[10];
  float* out = (float*)d_out;

  // workspace layout
  char* ws = (char*)d_ws;
  const size_t GIN_B  = (size_t)kT * kB * kG * sizeof(float);   // 256 MB
  const size_t WF_B   = (size_t)kG * kH * sizeof(short);        // 2 MB each
  const size_t DW_B   = (size_t)112 * kH * sizeof(short);       // 112 KB
  const size_t HF_B   = (size_t)2 * 4 * kKS * 32 * 16 * sizeof(short); // 128 KB
  unsigned* bar   = (unsigned*)ws;
  float*    gin   = (float*) (ws + 256);
  __bf16*   whhf  = (__bf16*)(ws + 256 + GIN_B);
  __bf16*   wihf  = (__bf16*)(ws + 256 + GIN_B + WF_B);
  __bf16*   decwf = (__bf16*)(ws + 256 + GIN_B + 2 * WF_B);
  __bf16*   hfrag = (__bf16*)(ws + 256 + GIN_B + 2 * WF_B + DW_B);
  __bf16*   hdfrg = (__bf16*)(ws + 256 + GIN_B + 2 * WF_B + DW_B + HF_B);

  prep_kernel<<<dim3(1024), dim3(256), 0, stream>>>(w_hh, w_ih, dec_w,
                                                    whhf, wihf, decwf, bar);
  gin_kernel<<<dim3(256, 32), dim3(256), 0, stream>>>(x, emb, wihf,
                                                      b_ih, b_hh, gin);
  rnn_kernel<<<dim3(kNBLK), dim3(256), 0, stream>>>(gin, whhf, decwf,
                                                    gamma, beta, dec_b, mask,
                                                    hfrag, hdfrg, out, bar);
}
